// FlashHyperbolicAttentionModule_43748536877559
// MI455X (gfx1250) — matrix operand order repaired and checked by reading, not compiled, not run on hardware
//
#include <hip/hip_runtime.h>
#include <hip/hip_bf16.h>
#include <math.h>

// Problem dims (fixed by reference)
#define BB 2
#define HH 16
#define NN 2048
#define DD 64
#define BH (BB*HH)

typedef __attribute__((ext_vector_type(16))) __bf16 v16bf;
typedef __attribute__((ext_vector_type(8)))  float  v8f;

union Frag { v16bf v; uint4 q[2]; };

struct MaskT   { static constexpr bool value = true;  };
struct NoMaskT { static constexpr bool value = false; };

// ---- CDNA5 async global->LDS path (guarded; falls back to reg-routed copy)
#if defined(__has_builtin)
#if __has_builtin(__builtin_amdgcn_global_load_async_to_lds_b128) && \
    __has_builtin(__builtin_amdgcn_s_wait_asynccnt)
#define HAVE_ASYNC_LDS 1
#endif
#endif

#ifdef HAVE_ASYNC_LDS
typedef int v4i_b128 __attribute__((vector_size(16)));
typedef __attribute__((address_space(1))) v4i_b128* gv4p;
typedef __attribute__((address_space(3))) v4i_b128* lv4p;
#endif

__device__ __forceinline__ void copy16_g2l(const unsigned short* g, unsigned short* l) {
#ifdef HAVE_ASYNC_LDS
    __builtin_amdgcn_global_load_async_to_lds_b128(
        (gv4p)(uintptr_t)g, (lv4p)(unsigned int)(uintptr_t)l, 0, 0);
#else
    *(uint4*)l = *(const uint4*)g;
#endif
}

template<int N>
__device__ __forceinline__ void wait_async() {
#ifdef HAVE_ASYNC_LDS
    __builtin_amdgcn_s_wait_asynccnt(N);
#endif
}

// f32 -> bf16 (round to nearest even)
__device__ __forceinline__ unsigned short f2bf(float f) {
    unsigned int x = __float_as_uint(f);
    unsigned int r = (x + 0x7fffu + ((x >> 16) & 1u)) >> 16;
    return (unsigned short)r;
}

// ---------------------------------------------------------------------------
// Prep 1: exp_map(q), exp_map(k) -> bf16 rows + f32 squared norms.
// ---------------------------------------------------------------------------
__global__ __launch_bounds__(256)
void prep_qk(const float* __restrict__ q, const float* __restrict__ k,
             unsigned short* __restrict__ qh, unsigned short* __restrict__ kh,
             float* __restrict__ qn2, float* __restrict__ kn2,
             const float* __restrict__ cp) {
    const float c      = fmaxf(cp[0], 1e-6f);
    const float sqrt_c = sqrtf(c);

    const int row = blockIdx.x * 16 + (threadIdx.x >> 4);  // 0 .. BH*NN-1
    const int l16 = threadIdx.x & 15;

    const float*    src  = (blockIdx.y == 0) ? q   : k;
    unsigned short* dsth = (blockIdx.y == 0) ? qh  : kh;
    float*          dstn = (blockIdx.y == 0) ? qn2 : kn2;

    const float4 x = *(const float4*)(src + (size_t)row * DD + l16 * 4);
    float s = x.x * x.x + x.y * x.y + x.z * x.z + x.w * x.w;
    s += __shfl_xor(s, 1); s += __shfl_xor(s, 2);
    s += __shfl_xor(s, 4); s += __shfl_xor(s, 8);

    const float norm  = fmaxf(sqrtf(s), 1e-6f);
    const float targ  = fminf(sqrt_c * norm, 15.0f);
    const float scale = tanhf(targ) / (sqrt_c * norm);

    ushort4 o;
    o.x = f2bf(x.x * scale); o.y = f2bf(x.y * scale);
    o.z = f2bf(x.z * scale); o.w = f2bf(x.w * scale);
    *(ushort4*)(dsth + (size_t)row * DD + l16 * 4) = o;

    if (l16 == 0) dstn[row] = s * scale * scale;   // ||x_hyp||^2
}

// ---------------------------------------------------------------------------
// Prep 2: V -> bf16 transposed [bh][d][n].
// ---------------------------------------------------------------------------
__global__ __launch_bounds__(256)
void prep_v(const float* __restrict__ v, unsigned short* __restrict__ vt) {
    const int idx = blockIdx.x * 256 + threadIdx.x;     // over BH*NN*DD
    const int d  = idx % DD;
    const int n  = (idx / DD) % NN;
    const int bh = idx / (DD * NN);
    vt[(size_t)bh * DD * NN + (size_t)d * NN + n] = f2bf(v[idx]);
}

// ---------------------------------------------------------------------------
// Flash hyperbolic attention: 4 waves x 16 query rows; 32-key blocks,
// double-buffered async global->LDS staging, exp2-domain online softmax.
// ---------------------------------------------------------------------------
#define KSTR 72   // LDS row stride, K tile (64 + pad) ushorts
#define VSTR 40   // LDS row stride, V^T tile (32 + pad) ushorts
#define PSTR 40   // LDS row stride, P tile (32 + pad) ushorts

// Stage one 32-key tile: K (32x64 bf16) row-major + V^T (64x32 bf16).
// 128 threads x 4 x 16B = 8KB (full coverage).
__device__ __forceinline__ void issue_tile(const unsigned short* kbase,
                                           const unsigned short* vbase,
                                           int key0, unsigned short* lk,
                                           unsigned short* lv, int tid) {
    const int kr = tid >> 2, ks = tid & 3;   // K: row 0..31, 4 threads/row
    const unsigned short* gk = kbase + (size_t)(key0 + kr) * DD;
    copy16_g2l(gk + ks * 8,        lk + kr * KSTR + ks * 8);
    copy16_g2l(gk + ks * 8 + 32,   lk + kr * KSTR + ks * 8 + 32);
    const int dv = tid >> 1, vs = tid & 1;   // V^T: d-row 0..63, 2 threads/row
    const unsigned short* gv = vbase + (size_t)dv * NN + key0;
    copy16_g2l(gv + vs * 8,        lv + dv * VSTR + vs * 8);
    copy16_g2l(gv + vs * 8 + 16,   lv + dv * VSTR + vs * 8 + 16);
}

__global__ __launch_bounds__(128)
void attn(const unsigned short* __restrict__ qh, const unsigned short* __restrict__ kh,
          const unsigned short* __restrict__ vt, const float* __restrict__ qn2,
          const float* __restrict__ kn2, const float* __restrict__ cp,
          const float* __restrict__ betap, float* __restrict__ out) {
    __shared__ __align__(16) unsigned short lds_k[2][32 * KSTR];  // [key][d]
    __shared__ __align__(16) unsigned short lds_v[2][64 * VSTR];  // V^T [d][key]
    __shared__ __align__(16) unsigned short lds_p[4][16 * PSTR];  // per-wave P [m][key]

    const int bh    = blockIdx.y;
    const int mblk  = blockIdx.x * 64;
    const int tid   = threadIdx.x;
    const int wave  = tid >> 5;
    const int lane  = tid & 31;
    const int lane16 = lane & 15;
    const int half   = lane >> 4;
    const int mbase  = mblk + wave * 16;

    const float c       = fmaxf(cp[0], 1e-6f);
    const float sqrt_c  = sqrtf(c);
    const float nb2     = -betap[0] / sqrt_c;   // score scale in log2 units
    const float NEG_INF = -__builtin_huge_valf();

    const unsigned short* qrow_base = qh + (size_t)bh * NN * DD;
    const unsigned short* krow_base = kh + (size_t)bh * NN * DD;
    const unsigned short* vrow_base = vt + (size_t)bh * DD * NN;
    const float* qn2b = qn2 + (size_t)bh * NN;
    const float* kn2b = kn2 + (size_t)bh * NN;

    // ---- Q A-fragments (documented 16-bit A layout)
    const int koff = half * 8;
    Frag aq0, aq1;
    {
        const unsigned short* p = qrow_base + (size_t)(mbase + lane16) * DD;
        aq0.q[0] = *(const uint4*)(p + koff);
        aq0.q[1] = *(const uint4*)(p + koff + 16);
        aq1.q[0] = *(const uint4*)(p + 32 + koff);
        aq1.q[1] = *(const uint4*)(p + 32 + koff + 16);
    }

    // Row stats + hoisted reciprocal row factors: fa_r = 2c / (1 - c*qn2_r)
    float qn2v[8], farow[8], mrow[8], lrow[8];
#pragma unroll
    for (int r = 0; r < 8; ++r) {
        const float qr = qn2b[mbase + half * 8 + r];
        qn2v[r]  = qr;
        farow[r] = (2.0f * c) * __builtin_amdgcn_rcpf(fmaxf(1.0f - c * qr, 1e-6f));
        mrow[r]  = NEG_INF;
        lrow[r]  = 0.0f;
    }

    v8f o[4];
#pragma unroll
    for (int t = 0; t < 4; ++t) o[t] = (v8f){};

    const int nblocks = blockIdx.x * 2 + 2;
    issue_tile(krow_base, vrow_base, 0, lds_k[0], lds_v[0], tid);

    for (int kb = 0; kb < nblocks; ++kb) {
        const int key0 = kb * 32;
        const unsigned short* lk = lds_k[kb & 1];
        const unsigned short* lv = lds_v[kb & 1];

        // Prefetch next tile into the other buffer (safe: it was retired at the
        // end-barrier of iteration kb-1), then wait for the current tile only.
        if (kb + 1 < nblocks) {
            issue_tile(krow_base, vrow_base, key0 + 32,
                       lds_k[(kb + 1) & 1], lds_v[(kb + 1) & 1], tid);
            wait_async<4>();      // async loads complete in order
        } else {
            wait_async<0>();
        }
        __syncthreads();          // publish tile kb

        if (key0 <= mbase + 15) {
            // ---- S = Q K^T (16x32)
            v8f s0 = (v8f){}, s1 = (v8f){};
            const int dhalf = half * 8;
            Frag b;
            {
                const unsigned short* kr = lk + lane16 * KSTR;
                b.q[0] = *(const uint4*)(kr + dhalf);
                b.q[1] = *(const uint4*)(kr + dhalf + 16);
                s0 = __builtin_amdgcn_wmma_f32_16x16x32_bf16(false, aq0.v, false, b.v, (short)0, s0, false, false);
                b.q[0] = *(const uint4*)(kr + 32 + dhalf);
                b.q[1] = *(const uint4*)(kr + 32 + dhalf + 16);
                s0 = __builtin_amdgcn_wmma_f32_16x16x32_bf16(false, aq1.v, false, b.v, (short)0, s0, false, false);
            }
            {
                const unsigned short* kr = lk + (16 + lane16) * KSTR;
                b.q[0] = *(const uint4*)(kr + dhalf);
                b.q[1] = *(const uint4*)(kr + dhalf + 16);
                s1 = __builtin_amdgcn_wmma_f32_16x16x32_bf16(false, aq0.v, false, b.v, (short)0, s1, false, false);
                b.q[0] = *(const uint4*)(kr + 32 + dhalf);
                b.q[1] = *(const uint4*)(kr + 32 + dhalf + 16);
                s1 = __builtin_amdgcn_wmma_f32_16x16x32_bf16(false, aq1.v, false, b.v, (short)0, s1, false, false);
            }

            // ---- transform + (optional) causal mask + online softmax (exp2)
            const float kn2_0 = kn2b[key0 + lane16];
            const float kn2_1 = kn2b[key0 + 16 + lane16];
            const float ib0 = __builtin_amdgcn_rcpf(fmaxf(1.0f - c * kn2_0, 1e-6f));
            const float ib1 = __builtin_amdgcn_rcpf(fmaxf(1.0f - c * kn2_1, 1e-6f));
            const int col0 = key0 + lane16, col1 = key0 + 16 + lane16;
            unsigned short* pls = lds_p[wave];

            auto softmax_block = [&](auto maskc) {
                constexpr bool DOMASK = decltype(maskc)::value;
#pragma unroll
                for (int r = 0; r < 8; ++r) {
                    const float sum0 = qn2v[r] + kn2_0;
                    const float sum1 = qn2v[r] + kn2_1;
                    float d0 = fmaxf(fmaf(-2.0f, s0[r], sum0), 0.0f);
                    float d1 = fmaxf(fmaf(-2.0f, s1[r], sum1), 0.0f);
                    float a0 = fmaxf(fmaf(d0, farow[r] * ib0, 1.0f), 1.0f + 1e-6f);
                    float a1 = fmaxf(fmaf(d1, farow[r] * ib1, 1.0f), 1.0f + 1e-6f);
                    float y0 = a0 + __builtin_amdgcn_sqrtf(fmaf(a0, a0, -1.0f));
                    float y1 = a1 + __builtin_amdgcn_sqrtf(fmaf(a1, a1, -1.0f));
                    float t0 = nb2 * __builtin_amdgcn_logf(y0);   // log2
                    float t1 = nb2 * __builtin_amdgcn_logf(y1);
                    if constexpr (DOMASK) {
                        const int row = mbase + half * 8 + r;
                        if (col0 > row) t0 = NEG_INF;
                        if (col1 > row) t1 = NEG_INF;
                    }

                    float red = fmaxf(t0, t1);
                    red = fmaxf(red, __shfl_xor(red, 1));
                    red = fmaxf(red, __shfl_xor(red, 2));
                    red = fmaxf(red, __shfl_xor(red, 4));
                    red = fmaxf(red, __shfl_xor(red, 8));
                    const float mnew  = fmaxf(mrow[r], red);
                    const float alpha = __builtin_amdgcn_exp2f(mrow[r] - mnew);
                    const float p0 = __builtin_amdgcn_exp2f(t0 - mnew);
                    const float p1 = __builtin_amdgcn_exp2f(t1 - mnew);
                    float ps = p0 + p1;
                    ps += __shfl_xor(ps, 1); ps += __shfl_xor(ps, 2);
                    ps += __shfl_xor(ps, 4); ps += __shfl_xor(ps, 8);
                    lrow[r] = lrow[r] * alpha + ps;
                    mrow[r] = mnew;
                    o[0][r] *= alpha; o[1][r] *= alpha; o[2][r] *= alpha; o[3][r] *= alpha;

                    const int prow = half * 8 + r;
                    pls[prow * PSTR + lane16]      = f2bf(p0);
                    pls[prow * PSTR + 16 + lane16] = f2bf(p1);
                }
            };
            if (key0 + 31 <= mbase) softmax_block(NoMaskT{});   // interior block
            else                    softmax_block(MaskT{});     // diagonal block

            // ---- O += P V (A = P 16x32, same-wave LDS RAW ordering)
            Frag pa;
            {
                const unsigned short* pp = pls + lane16 * PSTR + koff;
                pa.q[0] = *(const uint4*)(pp);
                pa.q[1] = *(const uint4*)(pp + 16);
            }
            Frag vb;
#pragma unroll
            for (int t = 0; t < 4; ++t) {
                const unsigned short* vp = lv + (t * 16 + lane16) * VSTR + half * 8;
                vb.q[0] = *(const uint4*)(vp);
                vb.q[1] = *(const uint4*)(vp + 16);
                o[t] = __builtin_amdgcn_wmma_f32_16x16x32_bf16(false, pa.v, false, vb.v, (short)0, o[t], false, false);
            }
        }
        __syncthreads();          // retire tile kb
    }

    // ---- finalize: O / l
    float* ob = out + (size_t)bh * NN * DD;
#pragma unroll
    for (int r = 0; r < 8; ++r) {
        const float rl = __builtin_amdgcn_rcpf(lrow[r]);
        float* orow = ob + (size_t)(mbase + half * 8 + r) * DD;
        orow[lane16]      = o[0][r] * rl;
        orow[16 + lane16] = o[1][r] * rl;
        orow[32 + lane16] = o[2][r] * rl;
        orow[48 + lane16] = o[3][r] * rl;
    }
}

// ---------------------------------------------------------------------------
extern "C" void kernel_launch(void* const* d_in, const int* in_sizes, int n_in,
                              void* d_out, int out_size, void* d_ws, size_t ws_size,
                              hipStream_t stream) {
    (void)in_sizes; (void)n_in; (void)out_size; (void)ws_size;
    const float* q    = (const float*)d_in[0];
    const float* k    = (const float*)d_in[1];
    const float* v    = (const float*)d_in[2];
    const float* c    = (const float*)d_in[3];
    const float* beta = (const float*)d_in[4];
    float* out = (float*)d_out;

    const size_t elems = (size_t)BH * NN * DD;
    unsigned short* qh = (unsigned short*)d_ws;
    unsigned short* kh = qh + elems;
    unsigned short* vt = kh + elems;
    float* qn2 = (float*)(vt + elems);
    float* kn2 = qn2 + (size_t)BH * NN;

    prep_qk<<<dim3(BH * NN / 16, 2), 256, 0, stream>>>(q, k, qh, kh, qn2, kn2, c);
    prep_v<<<dim3((unsigned)(elems / 256)), 256, 0, stream>>>(v, vt);
    attn<<<dim3(NN / 64, BH), 128, 0, stream>>>(qh, kh, vt, qn2, kn2, c, beta, out);
}
